// SkipConnection_41412074668295
// MI455X (gfx1250) — matrix operand order repaired and checked by reading, not compiled, not run on hardware
//
#include <hip/hip_runtime.h>
#include <hip/hip_bf16.h>
#include <math.h>

// ---------------------------------------------------------------------------
// MI455X (gfx1250) multi-scale skip-connection module.
// Large GEMMs: v_wmma_f32_16x16x32_f16 (f16 in, f32 acc), double-buffered LDS
// staging with vectorized b128 loads; softmax/LN/GELU/dwconv in fp32 VALU.
// ---------------------------------------------------------------------------

typedef __attribute__((ext_vector_type(16))) _Float16 v16h;
typedef __attribute__((ext_vector_type(8)))  _Float16 v8h;
typedef __attribute__((ext_vector_type(4)))  _Float16 v4h;
typedef __attribute__((ext_vector_type(8)))  float    v8f;

#define LDSW 40  // halfs per LDS row (80 B -> 16-byte aligned rows)

// Stage one 64x32 A tile and one 32x64 B tile (stored transposed) into LDS.
template <int AF16, int BT>
__device__ __forceinline__ void stage_tile(
    _Float16 (* __restrict__ As)[LDSW], _Float16 (* __restrict__ Bs)[LDSW],
    const float* __restrict__ Af, const _Float16* __restrict__ Ah,
    long aoff, long ldA,
    const float* __restrict__ Bptr, long boff, long ldB,
    int m0, int n0, int k0, int M, int N, int K,
    bool a_al, bool b_al, bool m_full, bool n_full, int tid)
{
    const bool k_full = (k0 + 32 <= K);

    // ---- A tile (64 rows x 32 k) ---------------------------------------
    {
        const int row = tid >> 1, seg = tid & 1, kb = seg * 16;
        const int gm = m0 + row;
        if (m_full && k_full && a_al) {
            if (AF16) {
                const v8h* src = (const v8h*)(Ah + aoff + (long)gm * ldA + k0 + kb);
                *(v8h*)&As[row][kb]     = src[0];
                *(v8h*)&As[row][kb + 8] = src[1];
            } else {
                const float4* src = (const float4*)(Af + aoff + (long)gm * ldA + k0 + kb);
#pragma unroll
                for (int i = 0; i < 4; ++i) {
                    float4 f = src[i];
                    v4h h = {(_Float16)f.x, (_Float16)f.y, (_Float16)f.z, (_Float16)f.w};
                    *(v4h*)&As[row][kb + i * 4] = h;
                }
            }
        } else {
            const long base = aoff + (long)gm * ldA + k0;
#pragma unroll
            for (int i = 0; i < 16; ++i) {          // branchless: clamp + select
                int kk = kb + i;
                bool ok = (gm < M) && ((k0 + kk) < K);
                long off = ok ? (base + kk) : aoff;
                _Float16 v = AF16 ? Ah[off] : (_Float16)Af[off];
                As[row][kk] = ok ? v : (_Float16)0.f;
            }
        }
    }
    // ---- B tile -> Bs[n][k] --------------------------------------------
    if (!BT) {
        const int k = tid >> 2, seg = tid & 3, nb = seg * 16;
        if (n_full && k_full && b_al) {
            const float4* src = (const float4*)(Bptr + boff + (long)(k0 + k) * ldB + n0 + nb);
#pragma unroll
            for (int i = 0; i < 4; ++i) {
                float4 f = src[i];
                Bs[nb + i * 4 + 0][k] = (_Float16)f.x;
                Bs[nb + i * 4 + 1][k] = (_Float16)f.y;
                Bs[nb + i * 4 + 2][k] = (_Float16)f.z;
                Bs[nb + i * 4 + 3][k] = (_Float16)f.w;
            }
        } else {
            const long base = boff + (long)(k0 + k) * ldB + n0;
#pragma unroll
            for (int i = 0; i < 16; ++i) {
                int n = nb + i;
                bool ok = ((k0 + k) < K) && ((n0 + n) < N);
                long off = ok ? (base + n) : boff;
                _Float16 v = (_Float16)Bptr[off];
                Bs[n][k] = ok ? v : (_Float16)0.f;
            }
        }
    } else {
        const int n = tid >> 1, seg = tid & 1, kb = seg * 16;
        if (n_full && k_full && b_al) {
            const float4* src = (const float4*)(Bptr + boff + (long)(n0 + n) * ldB + k0 + kb);
#pragma unroll
            for (int i = 0; i < 4; ++i) {
                float4 f = src[i];
                v4h h = {(_Float16)f.x, (_Float16)f.y, (_Float16)f.z, (_Float16)f.w};
                *(v4h*)&Bs[n][kb + i * 4] = h;
            }
        } else {
            const long base = boff + (long)(n0 + n) * ldB + k0;
#pragma unroll
            for (int i = 0; i < 16; ++i) {
                int kk = kb + i;
                bool ok = ((k0 + kk) < K) && ((n0 + n) < N);
                long off = ok ? (base + kk) : boff;
                _Float16 v = (_Float16)Bptr[off];
                Bs[n][kk] = ok ? v : (_Float16)0.f;
            }
        }
    }
}

// Generic WMMA GEMM: C[M,N] = A[M,K] @ B[K,N] (+bias)(+residual)(+=C)
// A: f32 or f16 row-major (template AF16); B: f32 [K,N] or [N,K] (template BT).
template <int AF16, int BT>
__global__ __launch_bounds__(128)
void gemm_wmma_t(const void* __restrict__ Aptr, long ldA, long strideA,
                 const float* __restrict__ Bptr, long ldB, long strideB,
                 float* __restrict__ Cptr, long ldC, long strideC,
                 const float* __restrict__ bias, int bias_mode, // 0 none,1 row,2 col
                 const float* __restrict__ resid, long ldR, long strideR,
                 int M, int N, int K, int accumulate)
{
    __shared__ _Float16 As[2][64][LDSW];   // double buffered
    __shared__ _Float16 Bs[2][64][LDSW];

    const int tid  = threadIdx.x;
    const int wave = tid >> 5;
    const int lane = tid & 31;
    const int l16  = lane & 15;
    const int hi   = lane >> 4;         // 0 or 1
    const int m0   = blockIdx.y * 64;
    const int n0   = blockIdx.x * 64;
    const int b    = blockIdx.z;

    const float*    Af = (const float*)Aptr;
    const _Float16* Ah = (const _Float16*)Aptr;
    const long aoff = (long)b * strideA;
    const long boff = (long)b * strideB;
    const long coff = (long)b * strideC;

    const bool a_al = AF16 ? ((ldA & 7) == 0) : ((ldA & 3) == 0);
    const bool b_al = (ldB & 3) == 0;
    const bool m_full = (m0 + 64 <= M);
    const bool n_full = (n0 + 64 <= N);

    v8f zero = {0.f,0.f,0.f,0.f,0.f,0.f,0.f,0.f};
    v8f acc[4];
#pragma unroll
    for (int j = 0; j < 4; ++j) acc[j] = zero;

    int cur = 0;
    stage_tile<AF16, BT>(As[0], Bs[0], Af, Ah, aoff, ldA, Bptr, boff, ldB,
                         m0, n0, 0, M, N, K, a_al, b_al, m_full, n_full, tid);
    __syncthreads();

    for (int k0 = 0; k0 < K; k0 += 32) {
        if (k0 + 32 < K)   // prefetch next tile into the other buffer
            stage_tile<AF16, BT>(As[cur ^ 1], Bs[cur ^ 1], Af, Ah, aoff, ldA,
                                 Bptr, boff, ldB, m0, n0, k0 + 32, M, N, K,
                                 a_al, b_al, m_full, n_full, tid);

        // ---- fragments: A and B = split-8 runs -------------
        union { v16h v; v8h h[2]; } afrag;
        const v8h* arow = (const v8h*)&As[cur][wave * 16 + l16][0];
        afrag.h[0] = arow[hi];        // K = hi*8 .. hi*8+7
        afrag.h[1] = arow[hi + 2];    // K = 16+hi*8 ..
        union { v16h v; v8h h[2]; } bfr[4];
#pragma unroll
        for (int j = 0; j < 4; ++j) {
            const v8h* brow = (const v8h*)&Bs[cur][j * 16 + l16][0];
            bfr[j].h[0] = brow[hi];           // K = hi*8 .. hi*8+7
            bfr[j].h[1] = brow[hi + 2];       // K = 16+hi*8 ..
        }
#pragma unroll
        for (int j = 0; j < 4; ++j)
            acc[j] = __builtin_amdgcn_wmma_f32_16x16x32_f16(
                false, afrag.v, false, bfr[j].v, (short)0, acc[j], false, false);

        __syncthreads();
        cur ^= 1;
    }

    // ---- epilogue: C layout VGPR g -> M = g + hi*8, N = l16 -------------
#pragma unroll
    for (int j = 0; j < 4; ++j) {
#pragma unroll
        for (int g = 0; g < 8; ++g) {
            int row = m0 + wave * 16 + g + hi * 8;
            int col = n0 + j * 16 + l16;
            if (row < M && col < N) {
                float v = acc[j][g];
                long off = coff + (long)row * ldC + col;
                if (accumulate)          v += Cptr[off];
                if (bias_mode == 1)      v += bias[row];
                else if (bias_mode == 2) v += bias[col];
                if (resid)               v += resid[(long)b * strideR + (long)row * ldR + col];
                Cptr[off] = v;
            }
        }
    }
}

__global__ void zero_kernel(float* p, long n)
{
    long i = (long)blockIdx.x * 256 + threadIdx.x;
    if (i < n) p[i] = 0.f;
}

// softmax along a row of length rowlen, row pitch rowstride (in place)
__global__ void softmax_rows_kernel(float* __restrict__ data, int rowlen, int rowstride)
{
    float* p = data + (long)blockIdx.x * rowstride;
    int tid = threadIdx.x;
    __shared__ float red[256];
    float mx = -3.4e38f;
    for (int i = tid; i < rowlen; i += 256) mx = fmaxf(mx, p[i]);
    red[tid] = mx; __syncthreads();
    for (int s = 128; s > 0; s >>= 1) { if (tid < s) red[tid] = fmaxf(red[tid], red[tid + s]); __syncthreads(); }
    mx = red[0]; __syncthreads();
    float sum = 0.f;
    for (int i = tid; i < rowlen; i += 256) { float e = expf(p[i] - mx); p[i] = e; sum += e; }
    red[tid] = sum; __syncthreads();
    for (int s = 128; s > 0; s >>= 1) { if (tid < s) red[tid] += red[tid + s]; __syncthreads(); }
    float inv = 1.f / red[0];
    for (int i = tid; i < rowlen; i += 256) p[i] *= inv;
}

// channel-block softmax of q + attention apply + transpose:
// enh[b,n,ci] = x[b,ci,n] + sum_k satt[b,k,v] * softmax_k(q[b,k*R+r,n]), ci=v*R+r
__global__ void enhance_kernel(const float* __restrict__ x, const float* __restrict__ q,
                               const float* __restrict__ satt_g, float* __restrict__ enh,
                               int Ci, int Nsp, int R)
{
    extern __shared__ float smem[];
    float* qs = smem;                 // [Ci][17]
    float* sa = smem + Ci * 17;       // [64*64]
    int b = blockIdx.y, tid = threadIdx.x;
    int n0 = blockIdx.x * 16;

    for (int i = tid; i < 4096; i += 256) sa[i] = satt_g[(long)b * 4096 + i];
    const float* qb = q + (long)b * Ci * Nsp;
    for (int idx = tid; idx < Ci * 16; idx += 256) {
        int c = idx >> 4, j = idx & 15, n = n0 + j;
        qs[c * 17 + j] = (n < Nsp) ? qb[(long)c * Nsp + n] : 0.f;
    }
    __syncthreads();
    if (tid < R * 16) {               // softmax over the 64 channel blocks
        int r = tid >> 4, j = tid & 15;
        float mx = -3.4e38f;
        for (int k = 0; k < 64; ++k) mx = fmaxf(mx, qs[(k * R + r) * 17 + j]);
        float s = 0.f;
        for (int k = 0; k < 64; ++k) { float e = expf(qs[(k * R + r) * 17 + j] - mx); qs[(k * R + r) * 17 + j] = e; s += e; }
        float inv = 1.f / s;
        for (int k = 0; k < 64; ++k) qs[(k * R + r) * 17 + j] *= inv;
    }
    __syncthreads();
    const float* xb = x + (long)b * Ci * Nsp;
    float* eb = enh + (long)b * Nsp * Ci;
    for (int idx = tid; idx < Ci * 16; idx += 256) {
        int ci = idx % Ci, j = idx / Ci, n = n0 + j;
        if (n < Nsp) {
            int v = ci / R, r = ci - v * R;
            float a = 0.f;
            for (int k = 0; k < 64; ++k) a += sa[k * 64 + v] * qs[(k * R + r) * 17 + j];
            eb[(long)n * Ci + ci] = xb[(long)ci * Nsp + n] + a;
        }
    }
}

// LayerNorm over last dim D; one wave per row
__global__ void ln_rows_kernel(const float* __restrict__ in, float* __restrict__ out,
                               const float* __restrict__ g, const float* __restrict__ bta,
                               long rows, int D)
{
    long row = (long)blockIdx.x * 8 + (threadIdx.x >> 5);
    int lane = threadIdx.x & 31;
    if (row >= rows) return;
    const float* p = in + row * (long)D;
    float s = 0.f, s2 = 0.f;
    for (int i = lane; i < D; i += 32) { float v = p[i]; s += v; s2 += v * v; }
    for (int m = 16; m > 0; m >>= 1) { s += __shfl_xor(s, m, 32); s2 += __shfl_xor(s2, m, 32); }
    float mean = s / D;
    float rstd = rsqrtf(s2 / D - mean * mean + 1e-5f);
    float* o = out + row * (long)D;
    for (int i = lane; i < D; i += 32) o[i] = (p[i] - mean) * rstd * g[i] + bta[i];
}

// fused depthwise 3x3 conv (SAME) + skip + LayerNorm + exact GELU -> f16
__global__ void dwconv_ln_gelu_kernel(const float* __restrict__ a,
                                      const float* __restrict__ wdw, const float* __restrict__ bdw,
                                      const float* __restrict__ g1, const float* __restrict__ b1,
                                      _Float16* __restrict__ h, int Hh, int Ww, int Hd)
{
    long gn = blockIdx.x;             // b*Nsp + n
    int Nsp = Hh * Ww;
    long bb = gn / Nsp;
    int nn = (int)(gn - bb * Nsp);
    int y = nn / Ww, x = nn - y * Ww;
    const float* ab = a + bb * (long)Nsp * Hd;
    int tid = threadIdx.x;

    float vals[8];
    int cnt = 0;
    float s = 0.f, s2 = 0.f;
    for (int hd = tid; hd < Hd; hd += 256) {
        float acc = bdw[hd] + ab[(long)nn * Hd + hd];   // conv bias + skip
#pragma unroll
        for (int ky = 0; ky < 3; ++ky) {
            int yy = y + ky - 1; if (yy < 0 || yy >= Hh) continue;
#pragma unroll
            for (int kx = 0; kx < 3; ++kx) {
                int xx = x + kx - 1; if (xx < 0 || xx >= Ww) continue;
                acc += ab[((long)yy * Ww + xx) * Hd + hd] * wdw[(ky * 3 + kx) * Hd + hd];
            }
        }
        vals[cnt++] = acc; s += acc; s2 += acc * acc;
    }
    __shared__ float rs[256], rq[256];
    rs[tid] = s; rq[tid] = s2; __syncthreads();
    for (int st = 128; st > 0; st >>= 1) {
        if (tid < st) { rs[tid] += rs[tid + st]; rq[tid] += rq[tid + st]; }
        __syncthreads();
    }
    float mean = rs[0] / Hd;
    float rstd = rsqrtf(rq[0] / Hd - mean * mean + 1e-5f);
    _Float16* hb = h + gn * (long)Hd;
    cnt = 0;
    for (int hd = tid; hd < Hd; hd += 256) {
        float t = (vals[cnt++] - mean) * rstd * g1[hd] + b1[hd];
        hb[hd] = (_Float16)(0.5f * t * (1.f + erff(t * 0.70710678f)));
    }
}

static void launch_gemm(hipStream_t stream, const void* A, int a_is_f16, long ldA, long strideA,
                        const float* B, int bT, long ldB, long strideB,
                        float* C, long ldC, long strideC,
                        const float* bias, int bias_mode,
                        const float* resid, long ldR, long strideR,
                        int M, int N, int K, int batch, int accumulate)
{
    dim3 grid((N + 63) / 64, (M + 63) / 64, batch);
    if (a_is_f16)
        gemm_wmma_t<1, 0><<<grid, 128, 0, stream>>>(A, ldA, strideA, B, ldB, strideB,
            C, ldC, strideC, bias, bias_mode, resid, ldR, strideR, M, N, K, accumulate);
    else if (bT)
        gemm_wmma_t<0, 1><<<grid, 128, 0, stream>>>(A, ldA, strideA, B, ldB, strideB,
            C, ldC, strideC, bias, bias_mode, resid, ldR, strideR, M, N, K, accumulate);
    else
        gemm_wmma_t<0, 0><<<grid, 128, 0, stream>>>(A, ldA, strideA, B, ldB, strideB,
            C, ldC, strideC, bias, bias_mode, resid, ldR, strideR, M, N, K, accumulate);
}

#define PRM(s, j) ((const float*)d_in[4 + (s) * 16 + (j)])

extern "C" void kernel_launch(void* const* d_in, const int* in_sizes, int n_in,
                              void* d_out, int out_size, void* d_ws, size_t ws_size,
                              hipStream_t stream)
{
    (void)in_sizes; (void)n_in; (void)out_size; (void)ws_size;
    const int Bn = 32;
    static const int CIs[4] = {64, 128, 320, 512};
    static const int HWs[4] = {56, 28, 14, 7};

    char* ws = (char*)d_ws;
    float*    sum_att = (float*)(ws + 0);           // 32*64*64
    float*    kbuf    = (float*)(ws + 524288);      // [B,64,Npad] / reused as qbuf
    float*    vbuf    = (float*)(ws + 26214400);    // [B,64,Npad] / reused as ybuf
    float*    enh     = (float*)(ws + 51904512);    // [B,Nsp,Ci]
    float*    abuf    = (float*)(ws + 77594624);    // [B,Nsp,4Ci]
    _Float16* hbuf    = (_Float16*)(ws + 180355072);// [B,Nsp,4Ci] f16
    float*    outp    = (float*)d_out;

    zero_kernel<<<(32 * 64 * 64 + 255) / 256, 256, 0, stream>>>(sum_att, 32L * 64 * 64);

    // ---------------- stage 1: sum_att = sum_i softmax(Wk x) (Wv x)^T ----
    for (int s = 0; s < 4; ++s) {
        int Ci = CIs[s], Hh = HWs[s], Nsp = Hh * Hh;
        int Np = (Nsp + 3) & ~3;       // padded pitch -> 16B-aligned rows
        const float* x = (const float*)d_in[s];
        launch_gemm(stream, PRM(s, 0), 0, Ci, 0, x, 0, Nsp, (long)Ci * Nsp,
                    kbuf, Np, 64L * Np, PRM(s, 1), 1, nullptr, 0, 0,
                    64, Nsp, Ci, Bn, 0);                               // k = Wk x + bk
        launch_gemm(stream, PRM(s, 2), 0, Ci, 0, x, 0, Nsp, (long)Ci * Nsp,
                    vbuf, Np, 64L * Np, PRM(s, 3), 1, nullptr, 0, 0,
                    64, Nsp, Ci, Bn, 0);                               // v = Wv x + bv
        softmax_rows_kernel<<<Bn * 64, 256, 0, stream>>>(kbuf, Nsp, Np);
        launch_gemm(stream, kbuf, 0, Np, 64L * Np, vbuf, 1, Np, 64L * Np,
                    sum_att, 64, 4096, nullptr, 0, nullptr, 0, 0,
                    64, 64, Nsp, Bn, 1);                               // sum_att += K V^T
    }

    // ---------------- stage 2: per-scale enhancement + MixFFN ------------
    long out_off = 0;
    for (int s = 0; s < 4; ++s) {
        int Ci = CIs[s], Hh = HWs[s], Nsp = Hh * Hh, Hd = 4 * Ci, R = Ci / 64;
        const float* x = (const float*)d_in[s];
        float* qbuf = kbuf;
        float* ybuf = vbuf;

        launch_gemm(stream, PRM(s, 4), 0, Ci, 0, x, 0, Nsp, (long)Ci * Nsp,
                    qbuf, Nsp, (long)Ci * Nsp, PRM(s, 5), 1, nullptr, 0, 0,
                    Ci, Nsp, Ci, Bn, 0);                               // q = Wq x + bq

        {   // softmax over channel blocks + attention apply + transpose
            dim3 grid((Nsp + 15) / 16, Bn);
            size_t sh = (size_t)(Ci * 17 + 4096) * sizeof(float);
            enhance_kernel<<<grid, 256, sh, stream>>>(x, qbuf, sum_att, enh, Ci, Nsp, R);
        }
        {   // y = LN(enh)
            long rows = (long)Bn * Nsp;
            ln_rows_kernel<<<(unsigned)((rows + 7) / 8), 256, 0, stream>>>(
                enh, ybuf, PRM(s, 6), PRM(s, 7), rows, Ci);
        }
        launch_gemm(stream, ybuf, 0, Ci, (long)Nsp * Ci, PRM(s, 8), 0, Hd, 0,
                    abuf, Hd, (long)Nsp * Hd, PRM(s, 9), 2, nullptr, 0, 0,
                    Nsp, Hd, Ci, Bn, 0);                               // a = y W1 + b1

        dwconv_ln_gelu_kernel<<<Bn * Nsp, 256, 0, stream>>>(
            abuf, PRM(s, 10), PRM(s, 11), PRM(s, 12), PRM(s, 13), hbuf, Hh, Hh, Hd);

        launch_gemm(stream, hbuf, 1, Hd, (long)Nsp * Hd, PRM(s, 14), 0, Ci, 0,
                    outp + out_off, Ci, (long)Nsp * Ci, PRM(s, 15), 2,
                    enh, Ci, (long)Nsp * Ci,
                    Nsp, Ci, Hd, Bn, 0);                               // out = h W2 + b2 + enh
        out_off += (long)Bn * Nsp * Ci;
    }
}
